// CoreferenceResolution_73598559584938
// MI455X (gfx1250) — compile-verified, matrix operand order checked by reading, not run on hardware
//
#include <hip/hip_runtime.h>
#include <hip/hip_bf16.h>

typedef __attribute__((ext_vector_type(16))) _Float16 v16h;
typedef __attribute__((ext_vector_type(8)))  _Float16 v8h;
typedef __attribute__((ext_vector_type(8)))  float    v8f;
typedef __attribute__((ext_vector_type(4)))  int      i32x4;

#define HIDDEN   768
#define META     25
#define IN_DIM   (2*HIDDEN + META)   /* 1561 */
#define K_PAD    1568                /* 49 * 32 */
#define KSTEP    32
#define NKSTEPS  (K_PAD / KSTEP)     /* 49 */
#define BATCH    2
#define N_MENT   2000
#define N_PAIRS  40000
#define M_TOTAL  (BATCH * N_PAIRS)   /* 80000 */
#define BLK_M    32
#define WAVE_N   96                  /* 6 x 16 cols per wave; 8 waves => 768 */
#define LDSPITCH 40                  /* padded halfs per LDS row (bank-spread) */

#define MENTION16_BYTES ((size_t)BATCH * N_MENT * HIDDEN * 2)      /* 6,144,000 */
#define W1T_BYTES       ((size_t)HIDDEN * K_PAD * 2)               /* 2,408,448 */

// ---- CDNA5 async global->LDS path (guarded; falls back to VGPR bounce) ----
#if defined(__has_builtin)
#  if __has_builtin(__builtin_amdgcn_global_load_async_to_lds_b128)
#    define HAVE_ASYNC 1
#  endif
#  if __has_builtin(__builtin_amdgcn_s_wait_asynccnt)
#    define HAVE_WAIT_ASYNC 1
#  endif
#endif
#ifndef HAVE_ASYNC
#  define HAVE_ASYNC 0
#endif
#ifndef HAVE_WAIT_ASYNC
#  define HAVE_WAIT_ASYNC 0
#endif

// Builtin signature (from clang diagnostic): param0 = int4 AS(1)*, param1 = int4 AS(3)*
typedef __attribute__((address_space(1))) i32x4 gvec_t;
typedef __attribute__((address_space(3))) i32x4 lvec_t;

// generic->AS1: same numeric address (global addresses are identity-mapped)
__device__ __forceinline__ gvec_t* to_global_v4(const void* p) {
  return (gvec_t*)(unsigned long long)p;
}
// generic->AS3: ISA aperture rule -> low 32 bits of a generic LDS address ARE the LDS offset
__device__ __forceinline__ lvec_t* to_lds_v4(void* p) {
  return (lvec_t*)(unsigned)(unsigned long long)p;
}

__device__ __forceinline__ void wait_async0() {
#if HAVE_WAIT_ASYNC
  __builtin_amdgcn_s_wait_asynccnt(0);
#else
  asm volatile("s_wait_asynccnt 0x0" ::: "memory");
#endif
}

// ---------------- preprocessing kernels ----------------

__global__ void cvt_f32_to_f16(const float* __restrict__ src,
                               _Float16* __restrict__ dst, int n) {
  int i = blockIdx.x * blockDim.x + threadIdx.x;
  if (i < n) dst[i] = (_Float16)src[i];
}

// W1 is [IN_DIM, HIDDEN] row-major; build W1t[HIDDEN][K_PAD] f16 (zero-padded K)
__global__ void build_w1t(const float* __restrict__ W1,
                          _Float16* __restrict__ W1t) {
  int i = blockIdx.x * blockDim.x + threadIdx.x;
  if (i >= HIDDEN * K_PAD) return;
  int n = i / K_PAD;
  int k = i - n * K_PAD;
  float v = (k < IN_DIM) ? W1[(size_t)k * HIDDEN + n] : 0.0f;
  W1t[i] = (_Float16)v;
}

// ed_table [300, META] -> ed_pad [300, 32] f16, zero-padded
__global__ void build_edpad(const float* __restrict__ ed,
                            _Float16* __restrict__ out) {
  int i = blockIdx.x * blockDim.x + threadIdx.x;
  if (i >= 300 * 32) return;
  int e = i >> 5, k = i & 31;
  out[i] = (_Float16)((k < META) ? ed[e * META + k] : 0.0f);
}

// ---------------- fused gather + GEMM(WMMA) + bias/ReLU + W2 dot ----------------

__global__ __launch_bounds__(256)
void coref_mlp_wmma(const _Float16* __restrict__ mention16,
                    const _Float16* __restrict__ w1t,
                    const _Float16* __restrict__ edpad,
                    const int*      __restrict__ pairs,
                    const int*      __restrict__ eds,
                    const float*    __restrict__ b1,
                    const float*    __restrict__ W2,
                    const float*    __restrict__ b2,
                    float*          __restrict__ out) {
  __shared__ __align__(16) _Float16 Alds[BLK_M][LDSPITCH];
  __shared__ __align__(16) _Float16 Blds[HIDDEN][LDSPITCH];
  __shared__ unsigned srcOff[3][BLK_M];   // element offsets (keeps loads GLOBAL)
  __shared__ float partial[BLK_M];

  const int t    = threadIdx.x;
  const int lane = t & 31;
  const int w    = t >> 5;        // wave id 0..7
  const int r0   = blockIdx.x * BLK_M;

  // per-row gather source offsets (one thread per output row)
  if (t < BLK_M) {
    int r  = r0 + t;
    int b  = r / N_PAIRS;
    int p  = r - b * N_PAIRS;
    int i0 = pairs[((size_t)b * N_PAIRS + p) * 2 + 0];
    int i1 = pairs[((size_t)b * N_PAIRS + p) * 2 + 1];
    int e  = eds[(size_t)b * N_PAIRS + p];
    srcOff[0][t] = (unsigned)(b * N_MENT + i0) * HIDDEN;
    srcOff[1][t] = (unsigned)(b * N_MENT + i1) * HIDDEN;
    srcOff[2][t] = (unsigned)e * 32;
    partial[t] = 0.0f;
  }

  v8f acc[2][6];
  #pragma unroll
  for (int mt = 0; mt < 2; ++mt)
    #pragma unroll
    for (int nt = 0; nt < 6; ++nt)
      acc[mt][nt] = (v8f){0.f, 0.f, 0.f, 0.f, 0.f, 0.f, 0.f, 0.f};

  const int nbase   = w * WAVE_N;
  const int halfsel = lane >> 4;    // 0: K {0..7,16..23}; 1: K {8..15,24..31}
  const int lrow    = lane & 15;
  const int kA      = halfsel * 8;  // half-element offset of low K group

  for (int step = 0; step < NKSTEPS; ++step) {
    const int kk = step * KSTEP;
    __syncthreads();  // previous iteration's fragment reads done; srcOff ready

    // ---- stage A tile: 32 rows x 32 halfs (gathered) ----
    if (t < 128) {
      int row = t >> 2, seg = t & 3;
      int idx = (kk >= 2 * HIDDEN) ? 2 : (kk >= HIDDEN ? 1 : 0);
      int off = kk - (idx == 2 ? 2 * HIDDEN : idx * HIDDEN);
      const _Float16* base = (idx == 2) ? edpad : mention16;
      const _Float16* s = base + srcOff[idx][row] + off + seg * 8;
#if HAVE_ASYNC
      __builtin_amdgcn_global_load_async_to_lds_b128(
          to_global_v4(s), to_lds_v4(&Alds[row][seg * 8]), 0, 0);
#else
      *(v8h*)&Alds[row][seg * 8] = *(const v8h*)s;
#endif
    }
    // ---- stage B tile: 768 cols x 32 halfs from pre-transposed W1t ----
    #pragma unroll
    for (int i = 0; i < 12; ++i) {
      int sidx = t + i * 256;             // 3072 16B segments
      int n = sidx >> 2, seg = sidx & 3;
      const _Float16* s = w1t + (size_t)n * K_PAD + kk + seg * 8;
#if HAVE_ASYNC
      __builtin_amdgcn_global_load_async_to_lds_b128(
          to_global_v4(s), to_lds_v4(&Blds[n][seg * 8]), 0, 0);
#else
      *(v8h*)&Blds[n][seg * 8] = *(const v8h*)s;
#endif
    }
#if HAVE_ASYNC
    wait_async0();
#endif
    __syncthreads();

    // ---- build fragments (ISA 16-bit A/B VGPR layout) ----
    v16h bfrag[6];
    #pragma unroll
    for (int nt = 0; nt < 6; ++nt) {
      int col = nbase + nt * 16 + lrow;
      v8h lo = *(const v8h*)&Blds[col][kA];
      v8h hi = *(const v8h*)&Blds[col][16 + kA];
      #pragma unroll
      for (int i = 0; i < 8; ++i) { bfrag[nt][i] = lo[i]; bfrag[nt][8 + i] = hi[i]; }
    }
    #pragma unroll
    for (int mt = 0; mt < 2; ++mt) {
      int row = mt * 16 + lrow;
      v8h lo = *(const v8h*)&Alds[row][kA];
      v8h hi = *(const v8h*)&Alds[row][16 + kA];
      v16h afrag;
      #pragma unroll
      for (int i = 0; i < 8; ++i) { afrag[i] = lo[i]; afrag[8 + i] = hi[i]; }
      #pragma unroll
      for (int nt = 0; nt < 6; ++nt) {
        acc[mt][nt] = __builtin_amdgcn_wmma_f32_16x16x32_f16(
            false, afrag, false, bfrag[nt], (short)0, acc[mt][nt], false, false);
      }
    }
  }

  // ---- fused epilogue: h = relu(acc + b1), rowsum += h * W2 ----
  float rowsum[2][8];
  #pragma unroll
  for (int mt = 0; mt < 2; ++mt)
    #pragma unroll
    for (int r = 0; r < 8; ++r) rowsum[mt][r] = 0.0f;

  #pragma unroll
  for (int nt = 0; nt < 6; ++nt) {
    int ncol = nbase + nt * 16 + lrow;   // C layout: N = lane%16
    float bb = b1[ncol];
    float ww = W2[ncol];
    #pragma unroll
    for (int mt = 0; mt < 2; ++mt)
      #pragma unroll
      for (int r = 0; r < 8; ++r) {
        float h = acc[mt][nt][r] + bb;
        h = h > 0.0f ? h : 0.0f;
        rowsum[mt][r] += h * ww;
      }
  }
  // reduce over the 16 lanes that share the same rows (xor masks stay in half)
  #pragma unroll
  for (int m = 8; m >= 1; m >>= 1) {
    #pragma unroll
    for (int mt = 0; mt < 2; ++mt)
      #pragma unroll
      for (int r = 0; r < 8; ++r)
        rowsum[mt][r] += __shfl_xor(rowsum[mt][r], m, 32);
  }
  if (lrow == 0) {
    #pragma unroll
    for (int mt = 0; mt < 2; ++mt)
      #pragma unroll
      for (int r = 0; r < 8; ++r) {
        int rowL = mt * 16 + halfsel * 8 + r;   // C layout: M = r + 8*(lane>=16)
        atomicAdd(&partial[rowL], rowsum[mt][r]);
      }
  }
  __syncthreads();
  if (t < BLK_M) out[r0 + t] = partial[t] + b2[0];
}

// ---------------- launcher ----------------

extern "C" void kernel_launch(void* const* d_in, const int* in_sizes, int n_in,
                              void* d_out, int out_size, void* d_ws, size_t ws_size,
                              hipStream_t stream) {
  const float* mention  = (const float*)d_in[0];   // [2,2000,768] f32
  const int*   pairs    = (const int*)  d_in[1];   // [2,40000,2]
  const int*   eds      = (const int*)  d_in[2];   // [2,40000]
  const float* ed_table = (const float*)d_in[3];   // [300,25]
  const float* W1       = (const float*)d_in[4];   // [1561,768]
  const float* b1       = (const float*)d_in[5];   // [768]
  const float* W2       = (const float*)d_in[6];   // [768,1]
  const float* b2       = (const float*)d_in[7];   // [1]
  float* out = (float*)d_out;                      // [2,40000] -> 80000 f32

  char* ws = (char*)d_ws;
  _Float16* mention16 = (_Float16*)ws;
  _Float16* w1t       = (_Float16*)(ws + MENTION16_BYTES);
  _Float16* edpad     = (_Float16*)(ws + MENTION16_BYTES + W1T_BYTES);

  {
    int n = BATCH * N_MENT * HIDDEN;                       // 3,072,000
    cvt_f32_to_f16<<<(n + 255) / 256, 256, 0, stream>>>(mention, mention16, n);
  }
  {
    int n = HIDDEN * K_PAD;                                // 1,204,224
    build_w1t<<<(n + 255) / 256, 256, 0, stream>>>(W1, w1t);
  }
  {
    int n = 300 * 32;                                      // 9,600
    build_edpad<<<(n + 255) / 256, 256, 0, stream>>>(ed_table, edpad);
  }

  coref_mlp_wmma<<<M_TOTAL / BLK_M, 256, 0, stream>>>(
      mention16, w1t, edpad, pairs, eds, b1, W2, b2, out);
}
